// Head_63067299774514
// MI455X (gfx1250) — matrix operand order repaired and checked by reading, not compiled, not run on hardware
//
#include <hip/hip_runtime.h>

typedef _Float16 v16h __attribute__((ext_vector_type(16)));
typedef _Float16 v8h  __attribute__((ext_vector_type(8)));
typedef float    v8f  __attribute__((ext_vector_type(8)));

#define B_   8
#define T_   2048
#define C_   1024
#define H_   64
#define BT_  (B_ * T_)

// Concatenate two 8-half vectors into a 16-half WMMA fragment register set.
__device__ __forceinline__ v16h cat8(v8h lo, v8h hi) {
  v16h r;
#pragma unroll
  for (int j = 0; j < 8; ++j) { r[j] = lo[j]; r[8 + j] = hi[j]; }
  return r;
}

__device__ __forceinline__ v8f zero8f() {
  v8f z;
#pragma unroll
  for (int i = 0; i < 8; ++i) z[i] = 0.0f;
  return z;
}

// Workspace layout (f16):
//   q  : [BT][H]      row-major        (A-side of q@k^T)
//   k  : [BT][H]      row-major        (contiguous per-lane B-frags, K dim = h)
//   vt : [B][H][T]    TRANSPOSED       (contiguous per-lane B-frags, K dim = s)
#define QOFF  ((size_t)0)
#define KOFF  ((size_t)BT_ * H_)
#define VOFF  ((size_t)2 * BT_ * H_)

// ---------------------------------------------------------------------------
// Kernel 1: fused QKV projection.  x:[BT,C] fp32  W*:[C,H] fp32
// Block: 256 threads (8 waves), 128 rows of x per block; blockIdx.y selects
// q/k/v. Per 32-wide K step: stage W chunk transposed (f16) in LDS, each wave
// runs 4 WMMAs (N = 64). V is written TRANSPOSED to [B][H][T] so the
// attention kernel needs no LDS transpose of V at all.
// ---------------------------------------------------------------------------
__global__ __launch_bounds__(256) void qkv_proj_kernel(
    const float* __restrict__ x, const float* __restrict__ Wq,
    const float* __restrict__ Wk, const float* __restrict__ Wv,
    _Float16* __restrict__ qkv)
{
  // WTs[h][kloc], row stride 40 halves (80B) -> 16B-aligned b128 reads,
  // conflict-free step-4 bank lattice over 64 banks.
  __shared__ __align__(16) _Float16 WTs[64 * 40];

  const int mat  = blockIdx.y;
  const float* W = (mat == 0) ? Wq : (mat == 1) ? Wk : Wv;

  const int tid   = threadIdx.x;
  const int wave  = tid >> 5;
  const int lane  = tid & 31;
  const int half_ = lane >> 4;   // which 16-lane half of the wave
  const int l15   = lane & 15;

  const int rbase = blockIdx.x * 128 + wave * 16;  // this wave's 16 rows
  const int arow  = rbase + l15;                   // A-frag: lane = M row
  const int koff  = half_ * 16;   // A layout: K groups {0-15}/{16-31}
  const int kseg  = half_ * 16;   // B layout: K groups {0-15}/{16-31}

  v8f acc[4];
#pragma unroll
  for (int nt = 0; nt < 4; ++nt) acc[nt] = zero8f();

  for (int kk = 0; kk < C_; kk += 32) {
    __syncthreads();  // all waves done reading previous chunk
    // Stage 32x64 fp32 W chunk -> f16 transposed in LDS (coalesced reads)
    for (int idx = tid; idx < 2048; idx += 256) {
      const int h  = idx & 63;
      const int kl = idx >> 6;
      WTs[h * 40 + kl] = (_Float16)W[(size_t)(kk + kl) * H_ + h];
    }
    __syncthreads();

    // A fragment: two 8-element contiguous runs of this row of x
    const float* xp = x + (size_t)arow * C_ + kk;
    float4 f0 = *(const float4*)(xp + koff);
    float4 f1 = *(const float4*)(xp + koff + 4);
    float4 f2 = *(const float4*)(xp + koff + 8);
    float4 f3 = *(const float4*)(xp + koff + 12);
    v16h a;
    a[0]  = (_Float16)f0.x; a[1]  = (_Float16)f0.y; a[2]  = (_Float16)f0.z; a[3]  = (_Float16)f0.w;
    a[4]  = (_Float16)f1.x; a[5]  = (_Float16)f1.y; a[6]  = (_Float16)f1.z; a[7]  = (_Float16)f1.w;
    a[8]  = (_Float16)f2.x; a[9]  = (_Float16)f2.y; a[10] = (_Float16)f2.z; a[11] = (_Float16)f2.w;
    a[12] = (_Float16)f3.x; a[13] = (_Float16)f3.y; a[14] = (_Float16)f3.z; a[15] = (_Float16)f3.w;

#pragma unroll
    for (int nt = 0; nt < 4; ++nt) {
      const _Float16* bp = &WTs[(nt * 16 + l15) * 40 + kseg];  // lane = N col
      v16h bb = cat8(*(const v8h*)bp, *(const v8h*)(bp + 8));
      acc[nt] = __builtin_amdgcn_wmma_f32_16x16x32_f16(
          false, a, false, bb, (short)0, acc[nt], false, false);
    }
  }

  // C/D layout: lane = N col, VGPR r holds M = r + 8*half_
  if (mat < 2) {
    _Float16* outp = qkv + (mat == 0 ? QOFF : KOFF);
#pragma unroll
    for (int nt = 0; nt < 4; ++nt)
#pragma unroll
      for (int r = 0; r < 8; ++r)
        outp[(size_t)(rbase + r + half_ * 8) * H_ + nt * 16 + l15] =
            (_Float16)acc[nt][r];
  } else {
    // V: write transposed [B][H][T]; blocks never straddle a batch boundary
    const int bidx = rbase / T_;
    const int t0   = rbase - bidx * T_;
    _Float16* vt   = qkv + VOFF + (size_t)bidx * H_ * T_;
#pragma unroll
    for (int nt = 0; nt < 4; ++nt)
#pragma unroll
      for (int r = 0; r < 8; ++r)
        vt[(size_t)(nt * 16 + l15) * T_ + t0 + r + half_ * 8] =
            (_Float16)acc[nt][r];
  }
}

// ---------------------------------------------------------------------------
// Kernel 2: causal flash attention, f16 operands / fp32 accumulate+output.
// Block = 128 threads (4 waves), wave w owns q rows [blk*64 + w*16, +16).
// Barrier-free: every operand fragment is a contiguous per-lane vector load
// (q/k row-major, V pre-transposed), so each wave streams only its own causal
// range (no ragged-edge idling). 4 waves per block share the K/V stream in L1.
// Per 32-col chunk: 4 score WMMAs, online softmax (shfl_xor half-wave row
// reductions), P transpose via per-wave LDS region, 4 P@V WMMAs.
// ---------------------------------------------------------------------------
__global__ __launch_bounds__(128) void attn_kernel(
    const _Float16* __restrict__ qkv, float* __restrict__ out)
{
  __shared__ __align__(16) _Float16 Psh[4 * 16 * 40];  // per-wave P tiles

  const int b      = blockIdx.y;
  const int tid    = threadIdx.x;
  const int wave   = tid >> 5;
  const int lane   = tid & 31;
  const int half_  = lane >> 4;
  const int l15    = lane & 15;
  const int koff   = half_ * 16;
  const int wtbase = blockIdx.x * 64 + wave * 16;   // this wave's q rows
  const int send   = wtbase + 16;                   // causal bound (exclusive)

  const _Float16* q  = qkv + QOFF + (size_t)b * T_ * H_;
  const _Float16* k  = qkv + KOFF + (size_t)b * T_ * H_;
  const _Float16* vt = qkv + VOFF + (size_t)b * H_ * T_;
  _Float16* Pw       = Psh + wave * 16 * 40;

  // Q fragments (M=16 rows, K=64 head dim -> two 16x32 A-frags), kept in regs
  v16h aq[2];
#pragma unroll
  for (int c = 0; c < 2; ++c) {
    const _Float16* qp = q + (size_t)(wtbase + l15) * H_ + c * 32;
    aq[c] = cat8(*(const v8h*)(qp + koff), *(const v8h*)(qp + koff + 8));
  }

  const float NEG_INF = -__builtin_inff();
  float mrow[8], lrow[8];
#pragma unroll
  for (int r = 0; r < 8; ++r) { mrow[r] = NEG_INF; lrow[r] = 0.0f; }
  v8f acc[4];
#pragma unroll
  for (int nt = 0; nt < 4; ++nt) acc[nt] = zero8f();

  for (int sbase = 0; sbase < send; sbase += 32) {
    // Prefetch next chunk of K rows and transposed-V columns.
    if (sbase + 32 < send) {
      __builtin_prefetch(k + (size_t)(sbase + 32 + lane) * H_, 0, 0);
      __builtin_prefetch(vt + (size_t)(lane * 2) * T_ + sbase + 32, 0, 0);
    }

    // ---- scores: two 16x16 subtiles, S = (Q @ K^T) / sqrt(H), causal mask
    v8f sf[2];
#pragma unroll
    for (int st = 0; st < 2; ++st) {
      v8f s = zero8f();
      const int srow = sbase + st * 16 + l15;  // lane = N column = s index
#pragma unroll
      for (int c = 0; c < 2; ++c) {
        const _Float16* kp = k + (size_t)srow * H_ + c * 32 + half_ * 16;
        v16h bb = cat8(*(const v8h*)kp, *(const v8h*)(kp + 8));
        s = __builtin_amdgcn_wmma_f32_16x16x32_f16(
            false, aq[c], false, bb, (short)0, s, false, false);
      }
#pragma unroll
      for (int r = 0; r < 8; ++r) {
        const int trow = wtbase + r + half_ * 8;
        float val = s[r] * 0.125f;            // 1/sqrt(64)
        if (srow > trow) val = NEG_INF;       // causal mask
        s[r] = val;
      }
      sf[st] = s;
    }

    // ---- online softmax: row max/sum across 16 lanes of each half-wave
    float rm[8], rs[8], alpha[8];
#pragma unroll
    for (int r = 0; r < 8; ++r) rm[r] = fmaxf(sf[0][r], sf[1][r]);
#pragma unroll
    for (int m = 1; m <= 8; m <<= 1)
#pragma unroll
      for (int r = 0; r < 8; ++r)
        rm[r] = fmaxf(rm[r], __shfl_xor(rm[r], m, 32));

#pragma unroll
    for (int r = 0; r < 8; ++r) {
      const float mnew = fmaxf(mrow[r], rm[r]);
      alpha[r] = __expf(mrow[r] - mnew);  // exp(-inf)=0 on first chunk
      mrow[r]  = mnew;
    }
#pragma unroll
    for (int st = 0; st < 2; ++st)
#pragma unroll
      for (int r = 0; r < 8; ++r)
        sf[st][r] = __expf(sf[st][r] - mrow[r]);  // masked -> exp(-inf)=0
#pragma unroll
    for (int r = 0; r < 8; ++r) rs[r] = sf[0][r] + sf[1][r];
#pragma unroll
    for (int m = 1; m <= 8; m <<= 1)
#pragma unroll
      for (int r = 0; r < 8; ++r)
        rs[r] += __shfl_xor(rs[r], m, 32);

#pragma unroll
    for (int r = 0; r < 8; ++r) lrow[r] = lrow[r] * alpha[r] + rs[r];
#pragma unroll
    for (int nt = 0; nt < 4; ++nt)
#pragma unroll
      for (int r = 0; r < 8; ++r) acc[nt][r] *= alpha[r];

    // ---- transpose P via per-wave LDS region (C-layout -> A-layout);
    //      same-wave DS ops are in-order, no barrier needed.
#pragma unroll
    for (int st = 0; st < 2; ++st)
#pragma unroll
      for (int r = 0; r < 8; ++r)
        Pw[(r + half_ * 8) * 40 + st * 16 + l15] = (_Float16)sf[st][r];

    const _Float16* pp = Pw + l15 * 40;
    v16h ap = cat8(*(const v8h*)(pp + koff), *(const v8h*)(pp + koff + 8));

    // ---- out += P @ V  (K = 32 s-cols, N = 64 head dims)
    //      V is pre-transposed: per-lane contiguous b128 loads, L1-shared.
#pragma unroll
    for (int nt = 0; nt < 4; ++nt) {
      const _Float16* vp = vt + (size_t)(nt * 16 + l15) * T_ + sbase + half_ * 16;
      v16h bv = cat8(*(const v8h*)vp, *(const v8h*)(vp + 8));
      acc[nt] = __builtin_amdgcn_wmma_f32_16x16x32_f16(
          false, ap, false, bv, (short)0, acc[nt], false, false);
    }
  }

  // ---- epilogue: out = acc / l  (fp32)
#pragma unroll
  for (int nt = 0; nt < 4; ++nt)
#pragma unroll
    for (int r = 0; r < 8; ++r)
      out[((size_t)b * T_ + (size_t)(wtbase + r + half_ * 8)) * H_ +
          nt * 16 + l15] = acc[nt][r] / lrow[r];
}

// ---------------------------------------------------------------------------
extern "C" void kernel_launch(void* const* d_in, const int* in_sizes, int n_in,
                              void* d_out, int out_size, void* d_ws, size_t ws_size,
                              hipStream_t stream) {
  const float* x  = (const float*)d_in[0];
  const float* Wq = (const float*)d_in[1];
  const float* Wk = (const float*)d_in[2];
  const float* Wv = (const float*)d_in[3];
  float* out      = (float*)d_out;

  // Workspace: q/k (row-major) + v (transposed), f16, 6 MB total ->
  // L2-resident on MI455X (192 MB L2).
  _Float16* qkv = (_Float16*)d_ws;

  dim3 gA(BT_ / 128, 3);
  qkv_proj_kernel<<<gA, dim3(256), 0, stream>>>(x, Wq, Wk, Wv, qkv);

  dim3 gB(T_ / 64, B_);
  attn_kernel<<<gB, dim3(128), 0, stream>>>(qkv, out);
}
